// gul_grs_user_model_11879879543067
// MI455X (gfx1250) — matrix operand order repaired and checked by reading, not compiled, not run on hardware
//
#include <hip/hip_runtime.h>
#include <hip/hip_bf16.h>

#define B_   32
#define L_   2048
#define D_   1024
#define H_   16
#define HD_  64
#define DFF_ 4096

typedef _Float16 half_t;
typedef __attribute__((ext_vector_type(16))) _Float16 v16h;
typedef __attribute__((ext_vector_type(2)))  _Float16 h2;
typedef __attribute__((ext_vector_type(8)))  float    v8f;

// ---------------------------------------------------------------------------
// WMMA fragment index helpers (CDNA5 ISA 7.12.2, wave32):
//  A (16x32 f16): lane L: m=L&15, g=L>>4; VGPR v holds halves K = k0(v,g), k0+1
//     with k0 = (v<4) ? 8g+2v : 16+8g+2(v-4)
//  B (32x16 f16): lane L: n=L&15, kb=16*(L>>4); this code instead puts K = kb+e in half element e of A and B alike
//  C/D (16x16 f32): lane L: n=L&15; VGPR j -> m = j + 8*(L>>4)
// ---------------------------------------------------------------------------
__device__ __forceinline__ int a_k0(int v, int g) {
    return (v < 4) ? (8 * g + 2 * v) : (16 + 8 * g + 2 * (v - 4));
}

// ---------------------------------------------------------------------------
// CDNA5 async global->LDS copy (ASYNCcnt path, ISA 15.18.3 op 98).
// LDS byte address = low 32 bits of flat shared pointer (ISA 10.2 aperture).
// ---------------------------------------------------------------------------
__device__ __forceinline__ void async_copy_b128(const void* gsrc, void* ldst) {
    unsigned lds = (unsigned)(size_t)ldst;
    unsigned long long ga = (unsigned long long)(size_t)gsrc;
    asm volatile("global_load_async_to_lds_b128 %0, %1, off"
                 :: "v"(lds), "v"(ga) : "memory");
}
__device__ __forceinline__ void async_wait0() {
    asm volatile("s_wait_asynccnt 0x0" ::: "memory");
}

// ---------------------------------------------------------------------------
// K1: qt[b][d][h] = (1/8) * sum_j Wk[d][h*64+j] * q[b][h*64+j]   (f16 out)
// ---------------------------------------------------------------------------
__global__ void qtilde_kernel(const float* __restrict__ q,
                              const float* __restrict__ Wk,
                              half_t* __restrict__ qt) {
    int b = blockIdx.y;
    int h = threadIdx.x & 15;
    int d = blockIdx.x * 16 + (threadIdx.x >> 4);
    const float* qrow = q + (size_t)b * D_ + h * HD_;
    const float* wrow = Wk + (size_t)d * D_ + h * HD_;
    float acc = 0.f;
#pragma unroll 8
    for (int j = 0; j < HD_; ++j) acc += qrow[j] * wrow[j];
    qt[((size_t)b * D_ + d) * H_ + h] = (half_t)(acc * 0.125f);
}

// ---------------------------------------------------------------------------
// K2: scores[b][h][l] = qt[b][h][:] . emb[ids[b][l]][:]   (masked to -1e9)
// Block: 256 thr = 8 waves; wave w owns 16 tokens.
// Hist tile staged f32 via async global->LDS; converted to f16 at frag build.
// ---------------------------------------------------------------------------
__global__ void __launch_bounds__(256) scores_kernel(
    const int* __restrict__ ids, const int* __restrict__ lens,
    const float* __restrict__ emb, const half_t* __restrict__ qt,
    float* __restrict__ scores) {
    __shared__ _Float16 sb[16 * 32];       // [h][k'] transposed qt tile
    __shared__ float    sa32[8][16 * 32];  // per-wave [m][k'] hist tile (f32)

    const int b    = blockIdx.y;
    const int l0   = blockIdx.x * 128;
    const int wave = threadIdx.x >> 5;
    const int lane = threadIdx.x & 31;
    const int nl   = lane & 15;
    const int g    = lane >> 4;

    // per-lane token row for A staging
    const int tok = l0 + wave * 16 + nl;
    const int row_id = ids[b * L_ + tok];
    const float* erow = emb + (size_t)row_id * D_;

    v8f c = {};
    for (int kk = 0; kk < D_; kk += 32) {
        __syncthreads();
        { // stage B: qt[b][kk..kk+32)[0..16) -> sb[h][k'] (transpose)
            int t = threadIdx.x;
            int dp = t >> 3;            // 0..31
            int hp = (t & 7) * 2;       // 0..14
            unsigned u = *(const unsigned*)(qt + ((size_t)b * D_ + kk + dp) * H_ + hp);
            h2 p = __builtin_bit_cast(h2, u);
            sb[hp * 32 + dp]       = p.x;
            sb[(hp + 1) * 32 + dp] = p.y;
        }
        // stage A (this wave): 16 rows x 32 f32 via async b128, 4 chunks/lane
#pragma unroll
        for (int i = 0; i < 4; ++i)
            async_copy_b128(erow + kk + g * 16 + i * 4,
                            &sa32[wave][nl * 32 + g * 16 + i * 4]);
        async_wait0();
        __syncthreads();
        v16h a, bf;
        const int kb = 16 * g;
#pragma unroll
        for (int v = 0; v < 8; ++v) {
            float2 fa = *(const float2*)&sa32[wave][nl * 32 + kb + 2 * v];
            a[2 * v] = (_Float16)fa.x; a[2 * v + 1] = (_Float16)fa.y;
            h2 pb = *(const h2*)&sb[nl * 32 + kb + 2 * v];
            bf[2 * v] = pb.x; bf[2 * v + 1] = pb.y;
        }
        c = __builtin_amdgcn_wmma_f32_16x16x32_f16(false, a, false, bf,
                                                   (short)0, c, false, false);
    }
    const int lenb = lens[b];
#pragma unroll
    for (int j = 0; j < 8; ++j) {
        int m = j + 8 * g;                      // token-in-wave-tile
        int l = l0 + wave * 16 + m;
        float s = (l < lenb) ? c[j] : -1e9f;
        scores[((size_t)b * H_ + nl) * L_ + l] = s;   // nl = head
    }
}

// ---------------------------------------------------------------------------
// K3: row softmax over L, write f16 attention weights.
// ---------------------------------------------------------------------------
__global__ void __launch_bounds__(256) softmax_kernel(
    const float* __restrict__ scores, half_t* __restrict__ attn) {
    __shared__ float red[8];
    const int row = blockIdx.x;                 // b*H + h
    const int t = threadIdx.x, lane = t & 31, wv = t >> 5;
    const float* src = scores + (size_t)row * L_;
    float v[8], m = -3.4e38f;
#pragma unroll
    for (int i = 0; i < 8; ++i) { v[i] = src[t + i * 256]; m = fmaxf(m, v[i]); }
#pragma unroll
    for (int o = 16; o; o >>= 1) m = fmaxf(m, __shfl_xor(m, o, 32));
    if (lane == 0) red[wv] = m;
    __syncthreads();
    m = red[0];
#pragma unroll
    for (int i = 1; i < 8; ++i) m = fmaxf(m, red[i]);
    float s = 0.f;
#pragma unroll
    for (int i = 0; i < 8; ++i) { v[i] = __expf(v[i] - m); s += v[i]; }
#pragma unroll
    for (int o = 16; o; o >>= 1) s += __shfl_xor(s, o, 32);
    __syncthreads();
    if (lane == 0) red[wv] = s;
    __syncthreads();
    s = 0.f;
#pragma unroll
    for (int i = 0; i < 8; ++i) s += red[i];
    const float inv = 1.f / s;
    half_t* dst = attn + (size_t)row * L_;
#pragma unroll
    for (int i = 0; i < 8; ++i) dst[t + i * 256] = (half_t)(v[i] * inv);
}

// ---------------------------------------------------------------------------
// K4: u[b][h][d] = sum_l attn[b][h][l] * emb[ids[b][l]][d]
// Block: 8 waves, wave w owns 16 d-columns. attn tile is already f16 in the
// exact [h][l] layout the A fragment wants -> pure async byte copy to LDS.
// ---------------------------------------------------------------------------
__global__ void __launch_bounds__(256) upool_kernel(
    const int* __restrict__ ids, const float* __restrict__ emb,
    const half_t* __restrict__ attn, float* __restrict__ u) {
    __shared__ _Float16 sa[16 * 32];  // [h][l']
    __shared__ int sids[32];

    const int b    = blockIdx.y;
    const int d0   = blockIdx.x * 128;
    const int wave = threadIdx.x >> 5;
    const int lane = threadIdx.x & 31;
    const int nl   = lane & 15;
    const int g    = lane >> 4;
    const int col  = d0 + wave * 16 + nl;

    v8f c = {};
    for (int kk = 0; kk < L_; kk += 32) {
        __syncthreads();
        { // stage A: attn[b][h][kk..kk+32) -> sa[h][k'] via async b128
            int t = threadIdx.x;
            if (t < 64) {
                int h = t >> 2, ch = t & 3;
                const half_t* gsrc = attn + ((size_t)b * H_ + h) * L_ + kk + ch * 8;
                async_copy_b128(gsrc, &sa[h * 32 + ch * 8]);
            }
            if (t < 32) sids[t] = ids[b * L_ + kk + t];
        }
        async_wait0();
        __syncthreads();
        v16h a, bf;
        const int kb = 16 * g;
#pragma unroll
        for (int e = 0; e < 16; ++e) {
            int id = sids[kb + e];
            bf[e] = (_Float16)emb[(size_t)id * D_ + col];
        }
#pragma unroll
        for (int v = 0; v < 8; ++v) {
            h2 pa = *(const h2*)&sa[nl * 32 + kb + 2 * v];
            a[2 * v] = pa.x; a[2 * v + 1] = pa.y;
        }
        c = __builtin_amdgcn_wmma_f32_16x16x32_f16(false, a, false, bf,
                                                   (short)0, c, false, false);
    }
#pragma unroll
    for (int j = 0; j < 8; ++j) {
        int h = j + 8 * g;      // A rows are heads
        u[((size_t)b * H_ + h) * D_ + col] = c[j];
    }
}

// ---------------------------------------------------------------------------
// K5: ctx[b][h*64+cc] = sum_d u[b][h][d] * Wv[d][h*64+cc]
// ---------------------------------------------------------------------------
__global__ void __launch_bounds__(256) ctx_kernel(
    const float* __restrict__ u, const float* __restrict__ Wv,
    float* __restrict__ ctx) {
    const int b = blockIdx.y;
    const int h = blockIdx.x * 4 + (threadIdx.x >> 6);
    const int cc = threadIdx.x & 63;
    const float* urow = u + ((size_t)b * H_ + h) * D_;
    const float* wcol = Wv + h * HD_ + cc;
    float acc = 0.f;
#pragma unroll 4
    for (int d = 0; d < D_; ++d) acc += urow[d] * wcol[(size_t)d * D_];
    ctx[(size_t)b * D_ + h * HD_ + cc] = acc;
}

// ---------------------------------------------------------------------------
// Generic small WMMA GEMM:  C = act(A[MxK] * B[KxN] + bias)
// grid (N/16, M/16), one wave per block.
// ---------------------------------------------------------------------------
__global__ void __launch_bounds__(32) gemm_wmma(
    const float* __restrict__ A, const float* __restrict__ Bm,
    const float* __restrict__ bias, float* __restrict__ C,
    int M, int N, int K, int relu) {
    const int lane = threadIdx.x & 31;
    const int nl = lane & 15, g = lane >> 4;
    const int m0 = blockIdx.y * 16, n0 = blockIdx.x * 16;
    v8f c = {};
    for (int kk = 0; kk < K; kk += 32) {
        v16h a, bf;
        const float* arow = A + (size_t)(m0 + nl) * K + kk;
#pragma unroll
        for (int v = 0; v < 8; ++v) {
            int k0 = 16 * g + 2 * v;
            a[2 * v]     = (_Float16)arow[k0];
            a[2 * v + 1] = (_Float16)arow[k0 + 1];
        }
        const int kb = 16 * g;
#pragma unroll
        for (int e = 0; e < 16; ++e)
            bf[e] = (_Float16)Bm[(size_t)(kk + kb + e) * N + n0 + nl];
        c = __builtin_amdgcn_wmma_f32_16x16x32_f16(false, a, false, bf,
                                                   (short)0, c, false, false);
    }
#pragma unroll
    for (int j = 0; j < 8; ++j) {
        int m = j + 8 * g;
        float v = c[j] + (bias ? bias[n0 + nl] : 0.f);
        if (relu) v = fmaxf(v, 0.f);
        C[(size_t)(m0 + m) * N + n0 + nl] = v;
    }
}

// ---------------------------------------------------------------------------
extern "C" void kernel_launch(void* const* d_in, const int* in_sizes, int n_in,
                              void* d_out, int out_size, void* d_ws, size_t ws_size,
                              hipStream_t stream) {
    const int*   ids   = (const int*)d_in[0];
    const int*   lens  = (const int*)d_in[1];
    const float* query = (const float*)d_in[2];
    const float* emb   = (const float*)d_in[3];
    const float* Wq    = (const float*)d_in[4];
    const float* Wk    = (const float*)d_in[5];
    const float* Wv    = (const float*)d_in[6];
    const float* Wo    = (const float*)d_in[7];
    const float* W1    = (const float*)d_in[8];
    const float* b1    = (const float*)d_in[9];
    const float* W2    = (const float*)d_in[10];
    const float* b2    = (const float*)d_in[11];
    float* out = (float*)d_out;

    // workspace carve-up (~10 MB)
    char* w = (char*)d_ws;
    float*  q      = (float*)w;                 w += (size_t)B_ * D_ * 4;        // 128 KB
    float*  scores = (float*)w;                 w += (size_t)B_ * H_ * L_ * 4;   // 4 MB
    float*  u      = (float*)w;                 w += (size_t)B_ * H_ * D_ * 4;   // 2 MB
    float*  ctx    = (float*)w;                 w += (size_t)B_ * D_ * 4;        // 128 KB
    float*  pooled = (float*)w;                 w += (size_t)B_ * D_ * 4;        // 128 KB
    float*  hbuf   = (float*)w;                 w += (size_t)B_ * DFF_ * 4;      // 512 KB
    half_t* qt     = (half_t*)w;                w += (size_t)B_ * D_ * H_ * 2;   // 1 MB
    half_t* attn   = (half_t*)w;                w += (size_t)B_ * H_ * L_ * 2;   // 2 MB

    // q = query @ Wq
    gemm_wmma<<<dim3(D_ / 16, B_ / 16), 32, 0, stream>>>(query, Wq, nullptr, q,
                                                         B_, D_, D_, 0);
    // q~ (head-restricted Wk^T q, scaled)
    qtilde_kernel<<<dim3(D_ / 16, B_), 256, 0, stream>>>(q, Wk, qt);
    // scores = q~ . hist  (fused gather, masked)
    scores_kernel<<<dim3(L_ / 128, B_), 256, 0, stream>>>(ids, lens, emb, qt, scores);
    // softmax -> f16 attention
    softmax_kernel<<<dim3(B_ * H_), 256, 0, stream>>>(scores, attn);
    // u = attn @ hist  (fused gather)
    upool_kernel<<<dim3(D_ / 128, B_), 256, 0, stream>>>(ids, emb, attn, u);
    // ctx = u @ Wv (head-restricted)
    ctx_kernel<<<dim3(H_ / 4, B_), 256, 0, stream>>>(u, Wv, ctx);
    // MLP head
    gemm_wmma<<<dim3(D_ / 16, B_ / 16), 32, 0, stream>>>(ctx, Wo, nullptr, pooled,
                                                         B_, D_, D_, 0);
    gemm_wmma<<<dim3(DFF_ / 16, B_ / 16), 32, 0, stream>>>(pooled, W1, b1, hbuf,
                                                           B_, DFF_, D_, 1);
    gemm_wmma<<<dim3(D_ / 16, B_ / 16), 32, 0, stream>>>(hbuf, W2, b2, out,
                                                         B_, D_, DFF_, 0);
}
